// ImplicitDecoder_26680336842945
// MI455X (gfx1250) — matrix operand order repaired and checked by reading, not compiled, not run on hardware
//
#include <hip/hip_runtime.h>
#include <hip/hip_bf16.h>

// ---------------------------------------------------------------------------
// ImplicitDecoder for MI455X (gfx1250, wave32, WMMA)
//   projection + bilinear feature gather + Fourier features  -> 16x256 f16 tile
//   MLP 256->64->64->64->1 via v_wmma_f32_16x16x32_f16 out of LDS
// ---------------------------------------------------------------------------

typedef __attribute__((ext_vector_type(16))) _Float16 v16h;
typedef __attribute__((ext_vector_type(8)))  _Float16 v8h;
typedef __attribute__((ext_vector_type(8)))  float    v8f;

#define FEAT_C   128
#define IMG_H    56
#define IMG_W    56
#define IMG_HW   (IMG_H * IMG_W)
#define MAP_M    64
#define HID      64
#define IN_DIM   (FEAT_C + 2 * MAP_M)        // 256
#define WAVES    8
#define BLOCK_T  (WAVES * 32)                // 256 threads
#define TPW      8                           // 16-point tiles per wave
#define PTS_PER_BLOCK (WAVES * TPW * 16)     // 1024 points / block

// A-fragment (16-bit A 16x32 layout): lane holds row (lane&15);
// lanes 0-15 take K offsets {0..7,16..23}, lanes 16-31 take {8..15,24..31}.
// -> two contiguous 16B LDS loads.
__device__ __forceinline__ v16h ld_afrag(const _Float16* p, int ko) {
  v8h lo = *(const v8h*)(p + ko);
  v8h hi = *(const v8h*)(p + ko + 16);
  v16h r;
#pragma unroll
  for (int i = 0; i < 8; ++i) { r[i] = lo[i]; r[i + 8] = hi[i]; }
  return r;
}

// B-fragment (16-bit B 32x16 layout): lane holds column (lane&15);
// lanes 0-15 take K {0..7,16..23}, lanes 16-31 take K {8..15,24..31}.
// Weights are stored transposed (Wt[n][k]) so this is two contiguous 16B runs.
__device__ __forceinline__ v16h ld_bfrag(const _Float16* p, int ko) {
  v8h lo = *(const v8h*)(p + ko);
  v8h hi = *(const v8h*)(p + ko + 16);
  v16h r;
#pragma unroll
  for (int i = 0; i < 8; ++i) { r[i] = lo[i]; r[i + 8] = hi[i]; }
  return r;
}

// One MLP layer on a 16-row tile: H_out(16xHID) = relu(A(16xK) @ Wt^T + bias),
// result written back (f16) into tile columns 0..63.
template <int KCHUNKS, int WSTRIDE>
__device__ __forceinline__ void mlp_layer(_Float16 (*tile)[IN_DIM],
                                          const _Float16* Wt,
                                          const float* bias,
                                          int pl, int hi16) {
  const int koffA = hi16 ? 8 : 0;
  const int koffB = hi16 ? 8 : 0;
  const int mbase = hi16 ? 8 : 0;
  const _Float16* arow = &tile[pl][0];
  v8f acc[4] = {};
#pragma unroll
  for (int t = 0; t < KCHUNKS; ++t) {
    v16h a = ld_afrag(arow, t * 32 + koffA);
#pragma unroll
    for (int j = 0; j < 4; ++j) {
      v16h b = ld_bfrag(Wt + (size_t)(j * 16 + pl) * WSTRIDE, t * 32 + koffB);
      acc[j] = __builtin_amdgcn_wmma_f32_16x16x32_f16(
          /*neg_a=*/false, a, /*neg_b=*/false, b,
          /*c_mod=*/(short)0, acc[j], /*reuse_a=*/false, /*reuse_b=*/false);
    }
  }
  // bias + relu in f32, store f16 back into tile cols 0..63 (C/D layout:
  // VGPR r -> M = r + mbase, column N = j*16 + (lane&15)).
#pragma unroll
  for (int j = 0; j < 4; ++j) {
    float bb = bias[j * 16 + pl];
#pragma unroll
    for (int r = 0; r < 8; ++r) {
      float hv = fmaxf(acc[j][r] + bb, 0.0f);
      tile[r + mbase][j * 16 + pl] = (_Float16)hv;
    }
  }
}

__global__ __launch_bounds__(BLOCK_T) void implicit_decoder_gfx1250(
    const float* __restrict__ feats,   // (B,128,56,56)
    const float* __restrict__ points,  // (B,N,3)
    const float* __restrict__ kmat,    // (B,3,3)
    const float* __restrict__ rtmat,   // (B,3,4)
    const float* __restrict__ Bg,      // (64,3)
    const float* __restrict__ W1, const float* __restrict__ b1,
    const float* __restrict__ W2, const float* __restrict__ b2,
    const float* __restrict__ W3, const float* __restrict__ b3,
    const float* __restrict__ W4, const float* __restrict__ b4,
    float* __restrict__ out, int totalPts, int nPerBatch) {
  __shared__ __align__(16) _Float16 sInp[WAVES][16][IN_DIM];  // 64 KB
  __shared__ __align__(16) _Float16 sW1t[HID][IN_DIM];        // 32 KB  Wt[n][k]
  __shared__ __align__(16) _Float16 sW2t[HID][HID];           // 8 KB
  __shared__ __align__(16) _Float16 sW3t[HID][HID];           // 8 KB
  __shared__ float sB1[HID], sB2[HID], sB3[HID], sW4[HID];
  __shared__ float sBg[MAP_M * 3];
  __shared__ float sB4;

  const int tid = threadIdx.x;

  // ---- stage weights into LDS, f16, transposed for contiguous B-fragments --
  for (int i = tid; i < HID * IN_DIM; i += BLOCK_T) {
    int n = i >> 8, k = i & (IN_DIM - 1);
    sW1t[n][k] = (_Float16)W1[k * HID + n];
  }
  for (int i = tid; i < HID * HID; i += BLOCK_T) {
    int n = i >> 6, k = i & 63;
    sW2t[n][k] = (_Float16)W2[k * HID + n];
    sW3t[n][k] = (_Float16)W3[k * HID + n];
  }
  if (tid < HID) {
    sB1[tid] = b1[tid]; sB2[tid] = b2[tid]; sB3[tid] = b3[tid];
    sW4[tid] = W4[tid];
  }
  if (tid < MAP_M * 3) sBg[tid] = Bg[tid];
  if (tid == 0) sB4 = b4[0];
  __syncthreads();

  const int wave = tid >> 5;
  const int lane = tid & 31;
  const int pl   = lane & 15;   // point-in-tile / fragment row-or-col index
  const int hi16 = lane >> 4;   // which half of the wave
  _Float16 (*tile)[IN_DIM] = sInp[wave];

  for (int t = 0; t < TPW; ++t) {
    const int tileBase = blockIdx.x * PTS_PER_BLOCK + (wave * TPW + t) * 16;
    if (tileBase + 16 > totalPts) continue;  // wave-uniform guard
    const int gp   = tileBase + pl;
    const int bidx = gp / nPerBatch;

    const float px = points[(size_t)gp * 3 + 0];
    const float py = points[(size_t)gp * 3 + 1];
    const float pz = points[(size_t)gp * 3 + 2];
    if (t + 1 < TPW)  // speculative prefetch of next tile's points
      __builtin_prefetch(&points[(size_t)(gp + 16) * 3], 0, 1);

    if (hi16 == 0) {
      // ---- projection + bilinear gather of 128 channels (lanes 0-15) ----
      const float* R = rtmat + bidx * 12;
      const float* K = kmat + bidx * 9;
      float cx = R[0] * px + R[1] * py + R[2]  * pz + R[3];
      float cy = R[4] * px + R[5] * py + R[6]  * pz + R[7];
      float cz = R[8] * px + R[9] * py + R[10] * pz + R[11];
      float ix = K[0] * cx + K[1] * cy + K[2] * cz;
      float iy = K[3] * cx + K[4] * cy + K[5] * cz;
      float iz = K[6] * cx + K[7] * cy + K[8] * cz;
      float invz = 1.0f / (iz + 1e-8f);
      // normalize->unnormalize chain cancels exactly: pixel coords = uv
      float u = ix * invz, v = iy * invz;
      float valid = (iz > 0.0f) ? 1.0f : 0.0f;
      float x0f = floorf(u), y0f = floorf(v);
      float fx = u - x0f, fy = v - y0f;
      int x0 = (int)x0f, y0 = (int)y0f, x1 = x0 + 1, y1 = y0 + 1;
      float m00 = (x0 >= 0 && x0 < IMG_W && y0 >= 0 && y0 < IMG_H) ? 1.f : 0.f;
      float m10 = (x1 >= 0 && x1 < IMG_W && y0 >= 0 && y0 < IMG_H) ? 1.f : 0.f;
      float m01 = (x0 >= 0 && x0 < IMG_W && y1 >= 0 && y1 < IMG_H) ? 1.f : 0.f;
      float m11 = (x1 >= 0 && x1 < IMG_W && y1 >= 0 && y1 < IMG_H) ? 1.f : 0.f;
      float w00 = (1.f - fx) * (1.f - fy) * valid * m00;
      float w10 = fx * (1.f - fy) * valid * m10;
      float w01 = (1.f - fx) * fy * valid * m01;
      float w11 = fx * fy * valid * m11;
      int cx0 = min(max(x0, 0), IMG_W - 1), cx1 = min(max(x1, 0), IMG_W - 1);
      int cy0 = min(max(y0, 0), IMG_H - 1), cy1 = min(max(y1, 0), IMG_H - 1);
      int a00 = cy0 * IMG_W + cx0, a10 = cy0 * IMG_W + cx1;
      int a01 = cy1 * IMG_W + cx0, a11 = cy1 * IMG_W + cx1;
      const float* fb = feats + (size_t)bidx * FEAT_C * IMG_HW;
#pragma unroll 4
      for (int c = 0; c < FEAT_C; ++c) {
        const float* fc = fb + (size_t)c * IMG_HW;
        float val = w00 * fc[a00] + w10 * fc[a10] +
                    w01 * fc[a01] + w11 * fc[a11];
        tile[pl][c] = (_Float16)val;
      }
    } else {
      // ---- Fourier features: sin/cos of 2*pi * (p . Bg[m]) (lanes 16-31) --
      const float TWO_PI = 6.28318530717958647692f;
#pragma unroll 4
      for (int m = 0; m < MAP_M; ++m) {
        float ang = TWO_PI * (px * sBg[m * 3 + 0] +
                              py * sBg[m * 3 + 1] +
                              pz * sBg[m * 3 + 2]);
        float s, c;
        __sincosf(ang, &s, &c);
        tile[pl][FEAT_C + m]         = (_Float16)s;
        tile[pl][FEAT_C + MAP_M + m] = (_Float16)c;
      }
    }
    // per-wave DS ops are in-order; same-array aliasing keeps store->load order
    __builtin_amdgcn_sched_barrier(0);

    // ---- MLP via WMMA: 256->64 (8 K-chunks), 64->64, 64->64 --------------
    mlp_layer<8, IN_DIM>(tile, &sW1t[0][0], sB1, pl, hi16);   // 32 wmma
    mlp_layer<2, HID>(tile, &sW2t[0][0], sB2, pl, hi16);      //  8 wmma
    mlp_layer<2, HID>(tile, &sW3t[0][0], sB3, pl, hi16);      //  8 wmma

    // ---- layer 4: 64 -> 1 dot product, lanes 0-15 (no relu) --------------
    if (hi16 == 0) {
      const _Float16* hrow = &tile[pl][0];
      float s = sB4;
#pragma unroll
      for (int i = 0; i < HID; ++i) s += (float)hrow[i] * sW4[i];
      out[gp] = s;
    }
    __builtin_amdgcn_sched_barrier(0);
  }
}

extern "C" void kernel_launch(void* const* d_in, const int* in_sizes, int n_in,
                              void* d_out, int out_size, void* d_ws, size_t ws_size,
                              hipStream_t stream) {
  const float* feats  = (const float*)d_in[0];
  const float* points = (const float*)d_in[1];
  const float* kmat   = (const float*)d_in[2];
  const float* rtmat  = (const float*)d_in[3];
  const float* Bg     = (const float*)d_in[4];
  const float* W1     = (const float*)d_in[5];
  const float* b1     = (const float*)d_in[6];
  const float* W2     = (const float*)d_in[7];
  const float* b2     = (const float*)d_in[8];
  const float* W3     = (const float*)d_in[9];
  const float* b3     = (const float*)d_in[10];
  const float* W4     = (const float*)d_in[11];
  const float* b4     = (const float*)d_in[12];
  float* out = (float*)d_out;

  const int totalPts  = in_sizes[1] / 3;                 // B * N = 524288
  const int nBatch    = in_sizes[3] / 12;                // rt is (B,3,4)
  const int nPerBatch = totalPts / nBatch;               // 65536
  const int nBlocks   = (totalPts + PTS_PER_BLOCK - 1) / PTS_PER_BLOCK;

  hipLaunchKernelGGL(implicit_decoder_gfx1250, dim3(nBlocks), dim3(BLOCK_T), 0,
                     stream, feats, points, kmat, rtmat, Bg, W1, b1, W2, b2,
                     W3, b3, W4, b4, out, totalPts, nPerBatch);
}
